// MultiHeadAttention_88210038326473
// MI455X (gfx1250) — matrix operand order repaired and checked by reading, not compiled, not run on hardware
//
#include <hip/hip_runtime.h>
#include <hip/hip_bf16.h>
#include <math.h>

// ---------------------------------------------------------------------------
// MHA forward for B=4, S=2048, HID=2048, H=16, DH=128 on gfx1250 (MI455X).
// bf16 WMMA (v_wmma_f32_16x16x32_bf16) for all matmuls, f32 accumulate.
// Flash-attention online softmax: never materializes S x S scores.
// K tiles staged into LDS by the Tensor Data Mover (tensor_load_to_lds,
// TENSORcnt) when the toolchain exposes the builtin (6-arg clang-23 form).
// ---------------------------------------------------------------------------

#define BATCH 4
#define SQ    2048
#define HIDN  2048
#define NH    16
#define DHD   128
#define NEGC  (-4294967295.0f)  /* -2^32 + 1, matches reference mask value */

typedef __attribute__((ext_vector_type(16))) __bf16 bf16x16;
typedef __attribute__((ext_vector_type(8)))  float  f32x8;
typedef __attribute__((ext_vector_type(4)))  unsigned int u32x4;
typedef __attribute__((ext_vector_type(8)))  int  i32x8;
typedef __attribute__((ext_vector_type(4)))  int  i32x4;

#if __has_builtin(__builtin_amdgcn_tensor_load_to_lds) && \
    __has_builtin(__builtin_amdgcn_s_wait_tensorcnt)
#define USE_TDM 1
#else
#define USE_TDM 0
#endif

union Frag {
  uint4   u[2];
  bf16x16 bf;
};

static __device__ __forceinline__ unsigned short f2bf(float f) {
  union { float f; unsigned int u; } c;
  c.f = f;
  unsigned int u = c.u;
  return (unsigned short)((u + 0x7FFFu + ((u >> 16) & 1u)) >> 16);  // RNE
}

// Pack two f32 -> packed bf16x2 dword.  Prefer hardware V_CVT_PK_BF16_F32.
static __device__ __forceinline__ unsigned pack2bf(float x, float y) {
#if __has_builtin(__builtin_amdgcn_cvt_pk_bf16_f32)
  auto r = __builtin_amdgcn_cvt_pk_bf16_f32(x, y);
  unsigned u;
  __builtin_memcpy(&u, &r, 4);
  return u;
#else
  return (unsigned)f2bf(x) | ((unsigned)f2bf(y) << 16);
#endif
}

static __device__ __forceinline__ f32x8 wmma_bf16(const Frag& a, const Frag& b, f32x8 c) {
  // v_wmma_f32_16x16x32_bf16: D = A(16x32) * B(32x16) + C(16x16 f32)
  return __builtin_amdgcn_wmma_f32_16x16x32_bf16(false, a.bf, false, b.bf,
                                                 (short)0, c, false, false);
}

// ---------------------------------------------------------------------------
// Pass 1: fused QKV projection GEMMs.  Out[m,n] = X[m,:] . W[n,:] + b[n]
// X: [8192,2048] f32, W: [2048,2048] f32 (torch Linear weight, K contiguous).
// Output stored bf16 (consumed by WMMA attention pass).
// blockIdx.z selects (queries,Wq)->Q, (keys,Wk)->K, (keys,Wv)->V.
// 128x128 tile / block, 8 waves, wave = 32x64 subtile, K-slab = 32.
// ---------------------------------------------------------------------------
__global__ __launch_bounds__(256) void qkv_gemm_kernel(
    const float* __restrict__ queries, const float* __restrict__ keys,
    const float* __restrict__ Wq, const float* __restrict__ bq,
    const float* __restrict__ Wk, const float* __restrict__ bk,
    const float* __restrict__ Wv, const float* __restrict__ bv,
    unsigned short* __restrict__ Qb, unsigned short* __restrict__ Kb,
    unsigned short* __restrict__ Vb)
{
  const int which = blockIdx.z;
  const float* __restrict__ X    = (which == 0) ? queries : keys;
  const float* __restrict__ W    = (which == 0) ? Wq : (which == 1) ? Wk : Wv;
  const float* __restrict__ bias = (which == 0) ? bq : (which == 1) ? bk : bv;
  unsigned short* __restrict__ Outp = (which == 0) ? Qb : (which == 1) ? Kb : Vb;

  const int m0 = blockIdx.y * 128;   // over M = B*S = 8192
  const int n0 = blockIdx.x * 128;   // over N = HID = 2048

  // bf16 staging tiles; row stride 40 ushorts = 80B (16B aligned, bank-spread)
  __shared__ unsigned short la[128][40];
  __shared__ unsigned short lb[128][40];

  const int t    = threadIdx.x;
  const int lane = t & 31;
  const int wave = t >> 5;          // 0..7
  const int l16  = lane & 15;
  const int lh   = lane >> 4;       // half-wave select
  const int wm   = (wave >> 1) * 32;
  const int wn   = (wave & 1) * 64;

  const f32x8 vzero = {};
  f32x8 acc[2][4];
#pragma unroll
  for (int mi = 0; mi < 2; ++mi)
#pragma unroll
    for (int ni = 0; ni < 4; ++ni) acc[mi][ni] = vzero;

  for (int k0 = 0; k0 < HIDN; k0 += 32) {
    __syncthreads();
    // Stage 128x32 f32 -> bf16 for both A (activations) and B (weights).
    // 1024 float4 loads per tile / 256 threads = 4 each.
#pragma unroll
    for (int i = 0; i < 4; ++i) {
      const int idx = t + i * 256;         // 0..1023
      const int row = idx >> 3;            // 0..127
      const int c4  = (idx & 7) << 2;      // 0,4,..,28
      float4 av = *(const float4*)(X + (size_t)(m0 + row) * HIDN + k0 + c4);
      float4 wv = *(const float4*)(W + (size_t)(n0 + row) * HIDN + k0 + c4);
      uint2 ap, bp;
      ap.x = pack2bf(av.x, av.y);
      ap.y = pack2bf(av.z, av.w);
      bp.x = pack2bf(wv.x, wv.y);
      bp.y = pack2bf(wv.z, wv.w);
      *(uint2*)&la[row][c4] = ap;
      *(uint2*)&lb[row][c4] = bp;
    }
    if (k0 + 32 < HIDN) {  // hint next K-slab -> global_prefetch_b8
      __builtin_prefetch(X + (size_t)(m0 + (t >> 1)) * HIDN + k0 + 32 + (t & 1) * 16, 0, 1);
      __builtin_prefetch(W + (size_t)(n0 + (t >> 1)) * HIDN + k0 + 32 + (t & 1) * 16, 0, 1);
    }
    __syncthreads();

    // A fragment (16x32 bf16): lane=M row; VGPR0-3 hold K=lh*8..+7,
    // VGPR4-7 hold K=16+lh*8..+7  (ISA 7.12.2, 16-bit A layout).
    Frag af[2];
#pragma unroll
    for (int mi = 0; mi < 2; ++mi) {
      const unsigned short* p = &la[wm + mi * 16 + l16][0];
      af[mi].u[0] = *(const uint4*)(p + lh * 8);
      af[mi].u[1] = *(const uint4*)(p + 16 + lh * 8);
    }
    // B fragment (32x16 bf16): lane=N col; VGPR0-3 hold K=lh*8..+7,
    // VGPR4-7 hold K=16+lh*8..+7 (same K order as A).
    Frag bfr[4];
#pragma unroll
    for (int ni = 0; ni < 4; ++ni) {
      const unsigned short* p = &lb[wn + ni * 16 + l16][0];
      bfr[ni].u[0] = *(const uint4*)(p + lh * 8);
      bfr[ni].u[1] = *(const uint4*)(p + 16 + lh * 8);
    }
#pragma unroll
    for (int mi = 0; mi < 2; ++mi)
#pragma unroll
      for (int ni = 0; ni < 4; ++ni)
        acc[mi][ni] = wmma_bf16(af[mi], bfr[ni], acc[mi][ni]);
  }

  // Epilogue: + bias, store bf16.  C layout: VGPR r -> row r + 8*lh, col l16.
#pragma unroll
  for (int ni = 0; ni < 4; ++ni) {
    const int nc = n0 + wn + ni * 16 + l16;
    const float bz = bias[nc];
#pragma unroll
    for (int mi = 0; mi < 2; ++mi) {
#pragma unroll
      for (int r = 0; r < 8; ++r) {
        const int mr = m0 + wm + mi * 16 + r + 8 * lh;
        Outp[(size_t)mr * HIDN + nc] = f2bf(acc[mi][ni][r] + bz);
      }
    }
  }
}

// ---------------------------------------------------------------------------
// Pass 1b: key padding mask.  kmask[b,s] = (sum_f keys[b,s,f] == 0) ? 0 : 1
// ---------------------------------------------------------------------------
__global__ __launch_bounds__(256) void kmask_kernel(const float* __restrict__ keys,
                                                    float* __restrict__ kmask)
{
  const int row  = blockIdx.x * 8 + (threadIdx.x >> 5);  // 0..8191
  const int lane = threadIdx.x & 31;
  const float* p = keys + (size_t)row * HIDN;
  float s = 0.0f;
  for (int i = lane; i < HIDN; i += 32) s += p[i];
#pragma unroll
  for (int d = 16; d >= 1; d >>= 1) s += __shfl_xor(s, d, 32);
  if (lane == 0) kmask[row] = (s == 0.0f) ? 0.0f : 1.0f;
}

// ---------------------------------------------------------------------------
// Pass 2: flash attention + residual.
// Block = (b, h, 128-query tile), 8 waves; each wave owns 16 query rows with
// Q fragments resident in VGPRs.  Per 32-key tile: K tile staged by the TDM
// (tensor_load_to_lds + s_wait_tensorcnt), V staged transposed by the block;
// scores = Q.K^T via 8 WMMAs, mask+scale, online softmax (shfl reductions),
// P staged via wave-private LDS (s_wait_dscnt) -> A fragment, PV via 8 WMMAs.
// ---------------------------------------------------------------------------
__global__ __launch_bounds__(256) void flash_attn_kernel(
    const unsigned short* __restrict__ Qb, const unsigned short* __restrict__ Kb,
    const unsigned short* __restrict__ Vb, const float* __restrict__ kmask,
    const float* __restrict__ queries, float* __restrict__ out)
{
  const int qt = blockIdx.x;   // 0..15 : query tile of 128
  const int h  = blockIdx.y;   // 0..15
  const int b  = blockIdx.z;   // 0..3
  const int t = threadIdx.x, lane = t & 31, wave = t >> 5;
  const int l16 = lane & 15, lh = lane >> 4;
  const int q0 = qt * 128;
  const int qw = q0 + wave * 16;     // this wave's 16 query rows

  __shared__ unsigned short kl[32][136];     // K tile  [key][dh],   272B rows
  __shared__ unsigned short vt[128][40];     // V^T tile [dh][key],   80B rows
  __shared__ unsigned short pl[8][16][40];   // per-wave P staging [q][key]

  // Q A-fragments (4 chunks of K=32 over DH=128), kept in VGPRs throughout.
  Frag qf[4];
  {
    const unsigned short* qp = Qb + (size_t)(b * SQ + qw + l16) * HIDN + h * DHD;
#pragma unroll
    for (int c = 0; c < 4; ++c) {
      qf[c].u[0] = *(const uint4*)(qp + c * 32 + lh * 8);
      qf[c].u[1] = *(const uint4*)(qp + c * 32 + 16 + lh * 8);
    }
  }

  const f32x8 vzero = {};
  f32x8 o[8];
#pragma unroll
  for (int i = 0; i < 8; ++i) o[i] = vzero;
  float mrow[8], lrow[8];
#pragma unroll
  for (int r = 0; r < 8; ++r) { mrow[r] = -INFINITY; lrow[r] = 0.0f; }

  const float scale = 0.02209708691207961f;  // 1/sqrt(2048), ref scales by HID
  const int kend = q0 + 128;                 // causal bound for this tile

  for (int kt = 0; kt < kend; kt += 32) {
    __syncthreads();

#if USE_TDM
    // --- K tile via Tensor Data Mover: one 2D descriptor, issued by wave 0.
    // D# group0: count=1 | lds_addr | global_addr | type=2 (ISA 8.3)
    // D# group1: data_size=2B, pad_enable (4 dwords every 64 dwords -> 272B
    // LDS row stride = kl[][136]), tensor_dim0=128, tile 128x32, stride 2048.
    if (wave == 0) {
      const unsigned lds_addr = (unsigned)(size_t)(void*)&kl[0][0];
      const unsigned long long ga =
          (unsigned long long)(size_t)(Kb + (size_t)(b * SQ + kt) * HIDN + h * DHD);
      u32x4 g0;
      g0[0] = 1u;                                        // count=1, user D#
      g0[1] = lds_addr;                                  // LDS byte address
      g0[2] = (unsigned)(ga & 0xFFFFFFFFu);              // global_addr[31:0]
      g0[3] = (unsigned)((ga >> 32) & 0x01FFFFFFu) | (2u << 30);  // [56:32]|type=2
      i32x8 g1;
      g1[0] = (int)((1u << 16) | (1u << 20) | (5u << 22) | (3u << 25));
      g1[1] = (int)(128u << 16);   // tensor_dim0[15:0] = 128 elements
      g1[2] = (int)(32u << 16);    // tensor_dim1[15:0] = 32 rows
      g1[3] = (int)(128u << 16);   // tile_dim0 = 128
      g1[4] = 32;                  // tile_dim1 = 32
      g1[5] = (int)HIDN;           // tensor_dim0_stride = 2048 elements
      g1[6] = 0;
      g1[7] = 0;
      i32x4 z4 = {};
      i32x8 z8 = {};
      // clang-23 / therock-10.0 form: 6 args (g0, g1, g2, g3, extra, cpol)
      __builtin_amdgcn_tensor_load_to_lds(g0, g1, z4, z4, z8, 0);
      __builtin_amdgcn_s_wait_tensorcnt(0);
    }
    // --- V tile staged transposed by all threads (TDM cannot transpose).
#pragma unroll
    for (int i = 0; i < 2; ++i) {
      const int idx = t + i * 256;      // 0..511 uint4 slots
      const int row = idx >> 4;         // local key 0..31
      const int c8  = (idx & 15) << 3;  // dh base 0,8,..,120
      const size_t g = (size_t)(b * SQ + kt + row) * HIDN + h * DHD + c8;
      union { uint4 q; unsigned short s[8]; } vv;
      vv.q = *(const uint4*)(Vb + g);
#pragma unroll
      for (int j = 0; j < 8; ++j) vt[c8 + j][row] = vv.s[j];
    }
#else
    // Fallback: stage K (as-is) and V (transposed) with vector loads.
#pragma unroll
    for (int i = 0; i < 2; ++i) {
      const int idx = t + i * 256;
      const int row = idx >> 4;
      const int c8  = (idx & 15) << 3;
      const size_t g = (size_t)(b * SQ + kt + row) * HIDN + h * DHD + c8;
      *(uint4*)&kl[row][c8] = *(const uint4*)(Kb + g);
      union { uint4 q; unsigned short s[8]; } vv;
      vv.q = *(const uint4*)(Vb + g);
#pragma unroll
      for (int j = 0; j < 8; ++j) vt[c8 + j][row] = vv.s[j];
    }
#endif
    __syncthreads();

    // scores[16q x 32k] = Q (16x128) . K^T : two 16-col subtiles, 4 K-chunks
    f32x8 sc[2];
    sc[0] = vzero; sc[1] = vzero;
#pragma unroll
    for (int sub = 0; sub < 2; ++sub) {
#pragma unroll
      for (int c = 0; c < 4; ++c) {
        Frag kf;  // B fragment of K^T: lane = key col, dh in A's K order
        const unsigned short* kp = &kl[sub * 16 + l16][c * 32];
        kf.u[0] = *(const uint4*)(kp + lh * 8);
        kf.u[1] = *(const uint4*)(kp + 16 + lh * 8);
        sc[sub] = wmma_bf16(qf[c], kf, sc[sub]);
      }
    }

    // mask + scale + online softmax (C layout: row = r + 8*lh, col = l16)
    const int kg0 = kt + l16, kg1 = kt + 16 + l16;
    const float km0 = kmask[b * SQ + kg0];
    const float km1 = kmask[b * SQ + kg1];
#pragma unroll
    for (int r = 0; r < 8; ++r) {
      const int qg = qw + r + 8 * lh;
      float v0 = sc[0][r] * scale;
      float v1 = sc[1][r] * scale;
      if (kg0 > qg || km0 == 0.0f) v0 = NEGC;
      if (kg1 > qg || km1 == 0.0f) v1 = NEGC;
      float mx = fmaxf(v0, v1);
#pragma unroll
      for (int d = 8; d >= 1; d >>= 1) mx = fmaxf(mx, __shfl_xor(mx, d, 32));
      const float mn = fmaxf(mrow[r], mx);
      const float alpha = __expf(mrow[r] - mn);
      const float p0 = __expf(v0 - mn);
      const float p1 = __expf(v1 - mn);
      float rs = p0 + p1;
#pragma unroll
      for (int d = 8; d >= 1; d >>= 1) rs += __shfl_xor(rs, d, 32);
      lrow[r] = lrow[r] * alpha + rs;
      mrow[r] = mn;
#pragma unroll
      for (int nt = 0; nt < 8; ++nt) o[nt][r] *= alpha;
      // stage P (bf16) to wave-private LDS in row-major [q][key]
      pl[wave][r + 8 * lh][l16]      = f2bf(p0);
      pl[wave][r + 8 * lh][16 + l16] = f2bf(p1);
    }

    // Wave-private LDS RAW (cross-lane): wait our DS stores before reloading.
    asm volatile("s_wait_dscnt 0x0" ::: "memory");

    Frag pf;  // A fragment of P (16q x 32k)
    {
      const unsigned short* pp = &pl[wave][l16][0];
      pf.u[0] = *(const uint4*)(pp + lh * 8);
      pf.u[1] = *(const uint4*)(pp + 16 + lh * 8);
    }
#pragma unroll
    for (int nt = 0; nt < 8; ++nt) {
      Frag vf;  // B fragment of V (32k x 16dh) from transposed staging
      const unsigned short* vp = &vt[nt * 16 + l16][0];
      vf.u[0] = *(const uint4*)(vp + lh * 8);
      vf.u[1] = *(const uint4*)(vp + 16 + lh * 8);
      o[nt] = wmma_bf16(pf, vf, o[nt]);
    }
  }

  // Normalize, add residual (out = softmax(QK^T)V + queries), write f32.
  float inv[8];
#pragma unroll
  for (int r = 0; r < 8; ++r) inv[r] = (lrow[r] > 0.0f) ? (1.0f / lrow[r]) : 0.0f;
#pragma unroll
  for (int nt = 0; nt < 8; ++nt) {
#pragma unroll
    for (int r = 0; r < 8; ++r) {
      const int qg = qw + r + 8 * lh;
      const size_t gi = (size_t)(b * SQ + qg) * HIDN + h * DHD + nt * 16 + l16;
      out[gi] = o[nt][r] * inv[r] + queries[gi];
    }
  }
}

// ---------------------------------------------------------------------------
extern "C" void kernel_launch(void* const* d_in, const int* in_sizes, int n_in,
                              void* d_out, int out_size, void* d_ws, size_t ws_size,
                              hipStream_t stream) {
  (void)in_sizes; (void)n_in; (void)out_size; (void)ws_size;
  const float* queries = (const float*)d_in[0];
  const float* keys    = (const float*)d_in[1];
  const float* Wq = (const float*)d_in[2];
  const float* bq = (const float*)d_in[3];
  const float* Wk = (const float*)d_in[4];
  const float* bk = (const float*)d_in[5];
  const float* Wv = (const float*)d_in[6];
  const float* bv = (const float*)d_in[7];
  float* out = (float*)d_out;

  // Workspace layout: Q,K,V bf16 [B*S, HID] + kmask f32 [B*S]  (~96 MB)
  const size_t elems = (size_t)BATCH * SQ * HIDN;
  unsigned short* Qb = (unsigned short*)d_ws;
  unsigned short* Kb = Qb + elems;
  unsigned short* Vb = Kb + elems;
  float* kmask = (float*)(Vb + elems);

  qkv_gemm_kernel<<<dim3(HIDN / 128, (BATCH * SQ) / 128, 3), 256, 0, stream>>>(
      queries, keys, Wq, bq, Wk, bk, Wv, bv, Qb, Kb, Vb);
  kmask_kernel<<<dim3((BATCH * SQ) / 8), 256, 0, stream>>>(keys, kmask);
  flash_attn_kernel<<<dim3(SQ / 128, NH, BATCH), 256, 0, stream>>>(
      Qb, Kb, Vb, kmask, queries, out);
}
